// RouteMoELayer_11201274708406
// MI455X (gfx1250) — matrix operand order repaired and checked by reading, not compiled, not run on hardware
//
#include <hip/hip_runtime.h>
#include <hip/hip_bf16.h>
#include <math.h>

// Problem sizes (match the reference)
#define B_    64
#define T_    32
#define H_    768
#define E_    8
#define NB_   2
#define DFF_  3072
#define NBEAM (B_*NB_)        // 128 beams

// d_out layout (floats, concatenated in reference return order)
#define OUT_ELEMS  (NBEAM*T_*H_)          // 3,145,728
#define BSCORE_OFF (OUT_ELEMS)            // +128
#define ROUTE_OFF  (BSCORE_OFF + NBEAM)   // +128 (int values stored as float)
#define BIDX_OFF   (ROUTE_OFF + NBEAM)    // +128
#define LOSS_OFF   (BIDX_OFF + NBEAM)     // +1

// workspace layout
#define WS_SCORES  0        // floats: 64*8 softmax scores
#define WS_ROUTE   512      // 128 ints (bit-stored in float slots)
#define WS_SCALE   640      // 128 per-beam output scales
#define WS_F16_OFF 4096     // byte offset of f16 mirror region
#define XN_   (B_*T_*H_)          // 1,572,864
#define W1N_  (E_*DFF_*H_)        // 18,874,368
#define W2N_  (E_*H_*DFF_)        // 18,874,368

typedef __attribute__((ext_vector_type(16))) _Float16 v16h;
typedef __attribute__((ext_vector_type(8)))  _Float16 v8h;
typedef __attribute__((ext_vector_type(8)))  float    v8f;

// ---- fragment helpers (CDNA5 wave32 WMMA layouts) ----

// B fragment (32x16): lane n holds 16 consecutive K of weight row n
// (lanes 0-15: K+0..15, lanes 16-31: K+16..31).
__device__ inline v16h load_b_frag_f32(const float* __restrict__ p) {
  const float4* q = (const float4*)p;
  float4 a0 = q[0], a1 = q[1], a2 = q[2], a3 = q[3];
  v16h r;
  r[0]=(_Float16)a0.x;  r[1]=(_Float16)a0.y;  r[2]=(_Float16)a0.z;  r[3]=(_Float16)a0.w;
  r[4]=(_Float16)a1.x;  r[5]=(_Float16)a1.y;  r[6]=(_Float16)a1.z;  r[7]=(_Float16)a1.w;
  r[8]=(_Float16)a2.x;  r[9]=(_Float16)a2.y;  r[10]=(_Float16)a2.z; r[11]=(_Float16)a2.w;
  r[12]=(_Float16)a3.x; r[13]=(_Float16)a3.y; r[14]=(_Float16)a3.z; r[15]=(_Float16)a3.w;
  return r;
}
__device__ inline v16h load_b_frag_f16(const _Float16* __restrict__ p) {
  return *(const v16h*)p;   // 32B contiguous -> two global_load_b128, no cvt
}

// A fragment (16x32): lane holds K sub-blocks [p0..p0+7] and [p1..p1+7] (called with p1 = p0 + 8: 16 consecutive K)
__device__ inline v16h load_a_frag_f32(const float* __restrict__ p0,
                                       const float* __restrict__ p1) {
  float4 a0 = ((const float4*)p0)[0], a1 = ((const float4*)p0)[1];
  float4 b0 = ((const float4*)p1)[0], b1 = ((const float4*)p1)[1];
  v16h r;
  r[0]=(_Float16)a0.x;  r[1]=(_Float16)a0.y;  r[2]=(_Float16)a0.z;  r[3]=(_Float16)a0.w;
  r[4]=(_Float16)a1.x;  r[5]=(_Float16)a1.y;  r[6]=(_Float16)a1.z;  r[7]=(_Float16)a1.w;
  r[8]=(_Float16)b0.x;  r[9]=(_Float16)b0.y;  r[10]=(_Float16)b0.z; r[11]=(_Float16)b0.w;
  r[12]=(_Float16)b1.x; r[13]=(_Float16)b1.y; r[14]=(_Float16)b1.z; r[15]=(_Float16)b1.w;
  return r;
}
__device__ inline v16h load_a_frag_f16(const _Float16* __restrict__ p0,
                                       const _Float16* __restrict__ p1) {
  v8h lo = *(const v8h*)p0, hi = *(const v8h*)p1;
  v16h r;
#pragma unroll
  for (int i = 0; i < 8; ++i) { r[i] = lo[i]; r[8 + i] = hi[i]; }
  return r;
}

// ==================== Kernel 0: fp32 -> f16 streaming convert ================
__global__ __launch_bounds__(256) void cvt_f16_kernel(const float* __restrict__ src,
                                                      _Float16* __restrict__ dst,
                                                      int n) {
  const int i = (blockIdx.x * 256 + threadIdx.x) * 8;
  if (i >= n) return;
  float4 a = ((const float4*)(src + i))[0];
  float4 b = ((const float4*)(src + i))[1];
  v8h r;
  r[0]=(_Float16)a.x; r[1]=(_Float16)a.y; r[2]=(_Float16)a.z; r[3]=(_Float16)a.w;
  r[4]=(_Float16)b.x; r[5]=(_Float16)b.y; r[6]=(_Float16)b.z; r[7]=(_Float16)b.w;
  *(v8h*)(dst + i) = r;
}

// ============================= Kernel 1: gating ==============================
__global__ __launch_bounds__(256) void gate_kernel(const float* __restrict__ x,
                                                   const float* __restrict__ gate_w,
                                                   float* __restrict__ out,
                                                   float* __restrict__ ws) {
  __shared__ float xa[H_];
  __shared__ float lg[E_];
  const int b   = blockIdx.x;
  const int tid = threadIdx.x;
  const float* xb = x + (size_t)b * T_ * H_;

  for (int h = tid; h < H_; h += 256) {
    float s = 0.f;
#pragma unroll
    for (int t = 0; t < T_; ++t) s += xb[t * H_ + h];
    xa[h] = s * (1.0f / T_);
  }
  __syncthreads();

  const int e = tid >> 5, lane = tid & 31;
  float p = 0.f;
  for (int h = lane; h < H_; h += 32) p += xa[h] * gate_w[e * H_ + h];
#pragma unroll
  for (int off = 16; off > 0; off >>= 1) p += __shfl_xor(p, off, 32);
  if (lane == 0) lg[e] = p;
  __syncthreads();

  if (tid == 0) {
    float m = lg[0];
#pragma unroll
    for (int i = 1; i < E_; ++i) m = fmaxf(m, lg[i]);
    float sc[E_]; float s = 0.f;
#pragma unroll
    for (int i = 0; i < E_; ++i) { sc[i] = expf(lg[i] - m); s += sc[i]; }
    const float inv = 1.f / s;
#pragma unroll
    for (int i = 0; i < E_; ++i) { sc[i] *= inv; ws[WS_SCORES + b * E_ + i] = sc[i]; }
    int i1 = 0;
#pragma unroll
    for (int i = 1; i < E_; ++i) if (sc[i] > sc[i1]) i1 = i;   // strict '>': lowest idx wins ties
    int i2 = (i1 == 0) ? 1 : 0;
#pragma unroll
    for (int i = 0; i < E_; ++i) { if (i != i1 && sc[i] > sc[i2]) i2 = i; }
    int idx[2] = { i1, i2 };
    int* route = (int*)(ws + WS_ROUTE);
#pragma unroll
    for (int k = 0; k < NB_; ++k) {
      const int beam = b * NB_ + k;
      out[BSCORE_OFF + beam] = sc[idx[k]];
      out[ROUTE_OFF  + beam] = (float)idx[k];
      out[BIDX_OFF   + beam] = (float)beam;
      route[beam] = idx[k];
      ws[WS_SCALE + beam] = sc[idx[k]];
    }
  }
}

// ========================= Kernel 2: importance loss =========================
__global__ void loss_kernel(const float* __restrict__ ws, float* __restrict__ out) {
  __shared__ float imp[E_];
  const int tid = threadIdx.x;
  if (tid < E_) {
    float s = 0.f;
    for (int b = 0; b < B_; ++b) s += ws[WS_SCORES + b * E_ + tid];
    imp[tid] = s;
  }
  __syncthreads();
  if (tid == 0) {
    float mean = 0.f;
#pragma unroll
    for (int i = 0; i < E_; ++i) mean += imp[i];
    mean *= (1.0f / E_);
    float v = 0.f;
#pragma unroll
    for (int i = 0; i < E_; ++i) { const float d = imp[i] - mean; v += d * d; }
    v *= (1.0f / (E_ - 1));             // ddof=1
    const float r = sqrtf(v) / mean;
    out[LOSS_OFF] = r * r;
  }
}

// ===================== Kernel 3: fused expert FFN (WMMA) =====================
#define CHUNK     512
#define HS_STRIDE (CHUNK + 8)   // halves; 260 dwords -> rows shift 4 LDS banks

template <bool PRECONV>
__global__ __launch_bounds__(256) void moe_ffn_kernel(const float* __restrict__ x,
                                                      const float* __restrict__ w1,
                                                      const float* __restrict__ b1,
                                                      const float* __restrict__ w2,
                                                      const float* __restrict__ b2,
                                                      const _Float16* __restrict__ xh,
                                                      const _Float16* __restrict__ w1h,
                                                      const _Float16* __restrict__ w2h,
                                                      const float* __restrict__ ws,
                                                      float* __restrict__ out) {
  __shared__ __align__(16) _Float16 hs[T_ * HS_STRIDE];   // 33,280 B

  const int beam = blockIdx.x;
  const int tid  = threadIdx.x;
  const int wave = tid >> 5;
  const int lane = tid & 31;

  const int*  route = (const int*)(ws + WS_ROUTE);
  const int   e     = route[beam];
  const float scale = ws[WS_SCALE + beam];
  const int   borig = beam >> 1;

  const float*    xb   = x   + (size_t)borig * T_ * H_;
  const _Float16* xbh  = xh  + (size_t)borig * T_ * H_;
  const float*    w1e  = w1  + (size_t)e * DFF_ * H_;
  const _Float16* w1eh = w1h + (size_t)e * DFF_ * H_;
  const float*    b1e  = b1  + (size_t)e * DFF_;
  const float*    w2e  = w2  + (size_t)e * H_ * DFF_;
  const _Float16* w2eh = w2h + (size_t)e * H_ * DFF_;
  const float*    b2e  = b2  + (size_t)e * H_;

  // per-lane fragment decomposition
  const int fm   = lane & 15;              // M (A/C row) or N (B/C col) index
  const int kh8  = (lane & 16) ? 8 : 0;    // unused (A and B frags both use kh16)
  const int kh16 = (lane & 16) ? 16 : 0;   // B-frag K sub-offset
  const int crow = (lane & 16) ? 8 : 0;    // C/D row offset for this lane half

  // GEMM2: persistent accumulators, 12 of the 96 output tiles per wave
  const int mt2     = wave >> 2;           // 0..1 -> T rows [0,16) / [16,32)
  const int nt2base = (wave & 3) * 12;     // 48 N-tiles of H=768 split 4 ways
  v8f acc2[12];
#pragma unroll
  for (int i = 0; i < 12; ++i) acc2[i] = (v8f){};

  // GEMM1: per chunk each wave owns 1 M-tile x 8 N-tiles
  const int mt1     = wave >> 2;
  const int nt1base = (wave & 3) * 8;

  for (int c = 0; c < DFF_; c += CHUNK) {
    // ---------- Phase 1: h_chunk = gelu(x @ w1_chunk^T + b1_chunk) ----------
    const float*    arow  = xb  + (size_t)(mt1 * 16 + fm) * H_;
    const _Float16* arowh = xbh + (size_t)(mt1 * 16 + fm) * H_;
#pragma unroll
    for (int g = 0; g < 2; ++g) {          // two groups of 4 tiles (reg pressure)
      v8f acc1[4];
#pragma unroll
      for (int i = 0; i < 4; ++i) acc1[i] = (v8f){};
      for (int k0 = 0; k0 < H_; k0 += 32) {
        const v16h a = PRECONV
            ? load_a_frag_f16(arowh + k0 + kh16, arowh + k0 + kh16 + 8)
            : load_a_frag_f32(arow  + k0 + kh16, arow  + k0 + kh16 + 8);
#pragma unroll
        for (int n = 0; n < 4; ++n) {
          const int fg = c + (nt1base + g * 4 + n) * 16 + fm;   // DFF row (B col)
          v16h bf;
          if (PRECONV) {
            const _Float16* brow = w1eh + (size_t)fg * H_ + k0 + kh16;
            if (n == 0) __builtin_prefetch(brow + 32, 0, 1);
            bf = load_b_frag_f16(brow);
          } else {
            const float* brow = w1e + (size_t)fg * H_ + k0 + kh16;
            if (n == 0) __builtin_prefetch(brow + 32, 0, 1);
            bf = load_b_frag_f32(brow);
          }
          acc1[n] = __builtin_amdgcn_wmma_f32_16x16x32_f16(
              false, a, false, bf, (short)0, acc1[n], false, false);
        }
      }
      // bias + exact-erf GELU + pack f16 into LDS h tile
#pragma unroll
      for (int n = 0; n < 4; ++n) {
        const int fl   = (nt1base + g * 4 + n) * 16 + fm;       // col within chunk
        const float bias = b1e[c + fl];
        const int  mrow = mt1 * 16 + crow;
        _Float16* col = hs + fl;
#pragma unroll
        for (int r = 0; r < 8; ++r) {
          const float v = acc1[n][r] + bias;
          const float gv = 0.5f * v * (1.0f + erff(v * 0.70710678118654752f));
          col[(mrow + r) * HS_STRIDE] = (_Float16)gv;
        }
      }
    }
    __syncthreads();

    // ---------- Phase 2: acc2 += h_chunk @ w2_chunk^T (K = this chunk) ------
    {
      const _Float16* hrow = hs + (size_t)(mt2 * 16 + fm) * HS_STRIDE;
      for (int k0 = 0; k0 < CHUNK; k0 += 32) {
        const v16h a = load_a_frag_f16(hrow + k0 + kh16, hrow + k0 + kh16 + 8);
#pragma unroll
        for (int n = 0; n < 12; ++n) {
          const int hg = (nt2base + n) * 16 + fm;               // H row (B col)
          v16h bf;
          if (PRECONV) {
            const _Float16* brow = w2eh + (size_t)hg * DFF_ + c + k0 + kh16;
            if (n == 0) __builtin_prefetch(brow + 32, 0, 1);
            bf = load_b_frag_f16(brow);
          } else {
            const float* brow = w2e + (size_t)hg * DFF_ + c + k0 + kh16;
            if (n == 0) __builtin_prefetch(brow + 32, 0, 1);
            bf = load_b_frag_f32(brow);
          }
          acc2[n] = __builtin_amdgcn_wmma_f32_16x16x32_f16(
              false, a, false, bf, (short)0, acc2[n], false, false);
        }
      }
    }
    __syncthreads();   // hs reused by next chunk's phase 1
  }

  // ---------------- epilogue: out = scale * (acc2 + b2) -----------------
  float* ob = out + (size_t)beam * T_ * H_;
  const int mrow = mt2 * 16 + crow;
#pragma unroll
  for (int n = 0; n < 12; ++n) {
    const int hg = (nt2base + n) * 16 + fm;
    const float bias = b2e[hg];
#pragma unroll
    for (int r = 0; r < 8; ++r) {
      ob[(size_t)(mrow + r) * H_ + hg] = scale * (acc2[n][r] + bias);
    }
  }
}

// ================================ launcher ==================================
extern "C" void kernel_launch(void* const* d_in, const int* in_sizes, int n_in,
                              void* d_out, int out_size, void* d_ws, size_t ws_size,
                              hipStream_t stream) {
  const float* x      = (const float*)d_in[0];
  const float* gate_w = (const float*)d_in[1];
  const float* w1     = (const float*)d_in[2];
  const float* b1     = (const float*)d_in[3];
  const float* w2     = (const float*)d_in[4];
  const float* b2     = (const float*)d_in[5];
  float* out = (float*)d_out;
  float* ws  = (float*)d_ws;

  hipLaunchKernelGGL(gate_kernel, dim3(B_), dim3(256), 0, stream, x, gate_w, out, ws);
  hipLaunchKernelGGL(loss_kernel, dim3(1), dim3(32), 0, stream, ws, out);

  const size_t f16_bytes = (size_t)(XN_ + W1N_ + W2N_) * sizeof(_Float16);
  const bool preconv = (ws_size >= WS_F16_OFF + f16_bytes);

  if (preconv) {
    _Float16* xh  = (_Float16*)((char*)d_ws + WS_F16_OFF);
    _Float16* w1h = xh + XN_;
    _Float16* w2h = w1h + W1N_;
    hipLaunchKernelGGL(cvt_f16_kernel, dim3((XN_  / 8 + 255) / 256), dim3(256), 0, stream, x,  xh,  XN_);
    hipLaunchKernelGGL(cvt_f16_kernel, dim3((W1N_ / 8 + 255) / 256), dim3(256), 0, stream, w1, w1h, W1N_);
    hipLaunchKernelGGL(cvt_f16_kernel, dim3((W2N_ / 8 + 255) / 256), dim3(256), 0, stream, w2, w2h, W2N_);
    hipLaunchKernelGGL((moe_ffn_kernel<true>), dim3(NBEAM), dim3(256), 0, stream,
                       x, w1, b1, w2, b2, xh, w1h, w2h, ws, out);
  } else {
    hipLaunchKernelGGL((moe_ffn_kernel<false>), dim3(NBEAM), dim3(256), 0, stream,
                       x, w1, b1, w2, b2, (const _Float16*)nullptr,
                       (const _Float16*)nullptr, (const _Float16*)nullptr, ws, out);
  }
}
